// BEVFormerBlock_61091614818919
// MI455X (gfx1250) — matrix operand order repaired and checked by reading, not compiled, not run on hardware
//
#include <hip/hip_runtime.h>

#define EMBED 256
#define NH 8
#define HD 32
#define SA_NP 4
#define CA_NP 8
#define NCAM 6
#define BEV_H 100
#define BEV_W 100
#define ZSIZE 8
#define DPIL 4
#define IMG_H 224
#define IMG_W 480
#define FH 28
#define FW 60
#define NQ (BEV_H * BEV_W)
#define LCAM (FH * FW)

typedef __attribute__((ext_vector_type(16))) _Float16 v16h;
typedef __attribute__((ext_vector_type(8)))  _Float16 v8h;
typedef __attribute__((ext_vector_type(8)))  float    v8f;

// ---------------------------------------------------------------------------
// LayerNorm over EMBED=256, one block per row; optional (h + bev_pos) output.
// ---------------------------------------------------------------------------
__global__ __launch_bounds__(EMBED)
void k_layernorm(const float* __restrict__ x, const float* __restrict__ g,
                 const float* __restrict__ b, const float* __restrict__ pos,
                 float* __restrict__ outh, float* __restrict__ outq) {
  int row = blockIdx.x, t = threadIdx.x;
  float v = x[row * EMBED + t];
  __shared__ float sb[EMBED];
  sb[t] = v; __syncthreads();
  for (int s = EMBED / 2; s > 0; s >>= 1) { if (t < s) sb[t] += sb[t + s]; __syncthreads(); }
  float mu = sb[0] * (1.0f / EMBED);
  __syncthreads();
  float d = v - mu;
  sb[t] = d * d; __syncthreads();
  for (int s = EMBED / 2; s > 0; s >>= 1) { if (t < s) sb[t] += sb[t + s]; __syncthreads(); }
  float inv = rsqrtf(sb[0] * (1.0f / EMBED) + 1e-5f);
  float y = d * inv * g[t] + b[t];
  outh[row * EMBED + t] = y;
  if (outq) outq[row * EMBED + t] = y + pos[row * EMBED + t];
}

// ---------------------------------------------------------------------------
// Pack weight W[K,N] (f32) -> Wt[N,K] (f16, transposed) for contiguous B frags.
// ---------------------------------------------------------------------------
__global__ __launch_bounds__(256)
void k_pack_wt(const float* __restrict__ W, _Float16* __restrict__ Wt, int K, int N) {
  int i = blockIdx.x * blockDim.x + threadIdx.x;
  if (i >= K * N) return;
  int k = i / N, n = i % N;
  Wt[n * K + k] = (_Float16)W[i];
}

// ---------------------------------------------------------------------------
// feats[n,l,c] = cam_feats[0,n,c,fh,fw] + camera_emb[n,c] + level_emb[c]
// ---------------------------------------------------------------------------
__global__ __launch_bounds__(256)
void k_feats_prep(const float* __restrict__ cf, const float* __restrict__ cam_emb,
                  const float* __restrict__ lvl_emb, float* __restrict__ feats) {
  int i = blockIdx.x * blockDim.x + threadIdx.x;
  if (i >= NCAM * LCAM * EMBED) return;
  int n = i / (LCAM * EMBED);
  int r = i - n * (LCAM * EMBED);
  int l = r / EMBED;
  int c = r % EMBED;
  feats[i] = cf[(n * EMBED + c) * LCAM + l] + cam_emb[n * EMBED + c] + lvl_emb[c];
}

// ---------------------------------------------------------------------------
// WMMA GEMM: out[M,N] = A[M,K](f32) x Wt[N,K](f16)^T + bias + res1 + res2,
// optional ReLU. One wave computes a 16 x (16*NT) strip: the A fragment is
// assembled once per K-step and reused by NT consecutive WMMAs.
// Requires M%16==0, N%(16*NT)==0, K%32==0 (true for all GEMMs here).
// ---------------------------------------------------------------------------
template <int NT>
__global__ __launch_bounds__(256)
void k_gemm_wmma(const float* __restrict__ A, const _Float16* __restrict__ Bt,
                 const float* __restrict__ bias, const float* __restrict__ res1,
                 const float* __restrict__ res2, float* __restrict__ out,
                 int M, int N, int K, int relu) {
  int wave = (blockIdx.x * blockDim.x + threadIdx.x) >> 5;
  int lane = threadIdx.x & 31;
  int stripsN = N / (16 * NT);
  int nstrips = (M >> 4) * stripsN;
  if (wave >= nstrips) return;         // whole-wave exit: EXEC stays all-ones
  int tm = wave / stripsN, ts = wave % stripsN;
  int half = lane >> 4;
  int l16  = lane & 15;

  const float* a_base = A + (size_t)((tm << 4) + l16) * K;
  const _Float16* b_base[NT];
#pragma unroll
  for (int t = 0; t < NT; t++)
    b_base[t] = Bt + (size_t)(ts * (16 * NT) + t * 16 + l16) * K;

  v8f acc[NT];
#pragma unroll
  for (int t = 0; t < NT; t++) acc[t] = (v8f){};

  for (int kk = 0; kk < K; kk += 32) {
    int k0 = kk + half * 8;            // A: two contiguous 8-float runs
    const float4* ap0 = reinterpret_cast<const float4*>(a_base + k0);
    const float4* ap1 = reinterpret_cast<const float4*>(a_base + k0 + 16);
    float4 a0 = ap0[0], a1 = ap0[1], a2 = ap1[0], a3 = ap1[1];
    if (kk + 32 < K)                   // gfx1250 global_prefetch_b8 for next slab
      __builtin_prefetch((const void*)(a_base + k0 + 32), 0, 0);
    v16h af;
    af[0]  = (_Float16)a0.x; af[1]  = (_Float16)a0.y; af[2]  = (_Float16)a0.z; af[3]  = (_Float16)a0.w;
    af[4]  = (_Float16)a1.x; af[5]  = (_Float16)a1.y; af[6]  = (_Float16)a1.z; af[7]  = (_Float16)a1.w;
    af[8]  = (_Float16)a2.x; af[9]  = (_Float16)a2.y; af[10] = (_Float16)a2.z; af[11] = (_Float16)a2.w;
    af[12] = (_Float16)a3.x; af[13] = (_Float16)a3.y; af[14] = (_Float16)a3.z; af[15] = (_Float16)a3.w;

    int kb = kk + half * 8;            // B: two 8-half runs (k, k+16) per tile
#pragma unroll
    for (int t = 0; t < NT; t++) {
      const v8h* bp = reinterpret_cast<const v8h*>(b_base[t] + kb);
      v8h b0 = bp[0], b1 = bp[2];
      v16h bf;
#pragma unroll
      for (int i = 0; i < 8; i++) { bf[i] = b0[i]; bf[i + 8] = b1[i]; }
      acc[t] = __builtin_amdgcn_wmma_f32_16x16x32_f16(false, af, false, bf,
                                                      (short)0, acc[t], false, false);
    }
  }

  int mB = (tm << 4) + (half << 3);
#pragma unroll
  for (int t = 0; t < NT; t++) {
    int nC = ts * (16 * NT) + t * 16 + l16;
    float bv = bias ? bias[nC] : 0.0f;
#pragma unroll
    for (int r = 0; r < 8; r++) {
      int m = mB + r;
      float v = acc[t][r] + bv;
      if (res1) v += res1[(size_t)m * N + nC];
      if (res2) v += res2[(size_t)m * N + nC];
      if (relu) v = fmaxf(v, 0.0f);
      out[(size_t)m * N + nC] = v;
    }
  }
}

// ---------------------------------------------------------------------------
// In-place softmax over last dim P (<=8), one thread per (q,head).
// ---------------------------------------------------------------------------
__global__ __launch_bounds__(256)
void k_softmax(float* __restrict__ a, int n, int P) {
  int i = blockIdx.x * blockDim.x + threadIdx.x;
  if (i >= n) return;
  float m = -1e30f;
  for (int p = 0; p < P; p++) m = fmaxf(m, a[i * P + p]);
  float e[8], s = 0.0f;
  for (int p = 0; p < P; p++) { e[p] = __expf(a[i * P + p] - m); s += e[p]; }
  float inv = 1.0f / s;
  for (int p = 0; p < P; p++) a[i * P + p] = e[p] * inv;
}

// ---------------------------------------------------------------------------
// Bilinear sample with border masking (matches reference valid*clip semantics).
// ---------------------------------------------------------------------------
__device__ __forceinline__
float bilin(const float* __restrict__ vbase, int H, int W, float x, float y, int ch) {
  float x0f = floorf(x), y0f = floorf(y);
  float fx = x - x0f, fy = y - y0f;
  int x0 = (int)x0f, y0 = (int)y0f;
  float w00 = (1.f - fx) * (1.f - fy), w10 = fx * (1.f - fy);
  float w01 = (1.f - fx) * fy,         w11 = fx * fy;
  float r = 0.0f;
  if (x0 >= 0     && x0 < W && y0 >= 0     && y0 < H) r += w00 * vbase[(size_t)(y0 * W + x0) * EMBED + ch];
  if (x0 + 1 >= 0 && x0 + 1 < W && y0 >= 0 && y0 < H) r += w10 * vbase[(size_t)(y0 * W + x0 + 1) * EMBED + ch];
  if (x0 >= 0     && x0 < W && y0 + 1 >= 0 && y0 + 1 < H) r += w01 * vbase[(size_t)((y0 + 1) * W + x0) * EMBED + ch];
  if (x0 + 1 >= 0 && x0 + 1 < W && y0 + 1 >= 0 && y0 + 1 < H) r += w11 * vbase[(size_t)((y0 + 1) * W + x0 + 1) * EMBED + ch];
  return r;
}

// ---------------------------------------------------------------------------
// BEV self deformable attention: block = query, wave = head, lane = channel.
// ---------------------------------------------------------------------------
__global__ __launch_bounds__(256)
void k_sa_deform(const float* __restrict__ v, const float* __restrict__ off,
                 const float* __restrict__ aw, float* __restrict__ out) {
  int q = blockIdx.x;
  int hh = threadIdx.x >> 5;
  int d  = threadIdx.x & 31;
  int iy = q / BEV_W, ix = q % BEV_W;
  float rx = (ix + 0.5f) / BEV_W, ry = (iy + 0.5f) / BEV_H;
  int ch = hh * HD + d;
  float acc = 0.0f;
  for (int p = 0; p < SA_NP; p++) {
    float ox = off[((size_t)(q * NH + hh) * SA_NP + p) * 2 + 0];
    float oy = off[((size_t)(q * NH + hh) * SA_NP + p) * 2 + 1];
    float x = (rx + ox / BEV_W) * BEV_W - 0.5f;
    float y = (ry + oy / BEV_H) * BEV_H - 0.5f;
    float a = aw[(size_t)(q * NH + hh) * SA_NP + p];
    acc += a * bilin(v, BEV_H, BEV_W, x, y, ch);
  }
  out[(size_t)q * EMBED + ch] = acc;
}

// ---------------------------------------------------------------------------
// Lidar->image projection: ref_cam[n,q,dl,2] and per-(n,q) hit mask.
// ---------------------------------------------------------------------------
__global__ __launch_bounds__(256)
void k_point_sampling(const float* __restrict__ l2i, float* __restrict__ ref_cam,
                      float* __restrict__ hit) {
  int i = blockIdx.x * blockDim.x + threadIdx.x;
  if (i >= NCAM * NQ) return;
  int n = i / NQ, q = i % NQ;
  int iy = q / BEV_W, ix = q % BEV_W;
  float X = ((ix + 0.5f) / BEV_W) * 100.0f - 50.0f;
  float Y = ((iy + 0.5f) / BEV_H) * 100.0f - 50.0f;
  const float* M = l2i + n * 16;
  int any = 0;
  for (int dl = 0; dl < DPIL; dl++) {
    float zr = (0.5f + dl * (7.0f / 3.0f)) / (float)ZSIZE;   // linspace(.5,7.5,4)/8
    float Z = zr * 20.0f - 10.0f;
    float p0 = M[0] * X + M[1] * Y + M[2] * Z + M[3];
    float p1 = M[4] * X + M[5] * Y + M[6] * Z + M[7];
    float p2 = M[8] * X + M[9] * Y + M[10] * Z + M[11];
    float denom = fmaxf(p2, 1e-5f);
    float px = (p0 / denom) / (float)IMG_W;
    float py = (p1 / denom) / (float)IMG_H;
    int m = (p2 > 1e-5f) && (px > 0.f) && (px < 1.f) && (py > 0.f) && (py < 1.f);
    any |= m;
    ref_cam[((size_t)(n * NQ + q) * DPIL + dl) * 2 + 0] = px;
    ref_cam[((size_t)(n * NQ + q) * DPIL + dl) * 2 + 1] = py;
  }
  hit[(size_t)n * NQ + q] = any ? 1.0f : 0.0f;
}

// ---------------------------------------------------------------------------
// Camera cross deformable attention fused with hit-masked camera reduction:
// slots[q,c] = sum_n hit[n,q]*deform_n / max(sum_n hit, 1)
// block = query, wave = head, lane = channel. Offsets/attn shared across cams.
// ---------------------------------------------------------------------------
__global__ __launch_bounds__(256)
void k_ca_deform(const float* __restrict__ vv, const float* __restrict__ off,
                 const float* __restrict__ aw, const float* __restrict__ ref_cam,
                 const float* __restrict__ hit, float* __restrict__ slots) {
  int q = blockIdx.x;
  int hh = threadIdx.x >> 5;
  int d  = threadIdx.x & 31;
  int ch = hh * HD + d;
  float acc = 0.0f, cnt = 0.0f;
  for (int n = 0; n < NCAM; n++) {
    float hv = hit[(size_t)n * NQ + q];
    cnt += hv;
    if (hv == 0.0f) continue;                 // uniform across wave (depends on n,q)
    const float* vb = vv + (size_t)n * LCAM * EMBED;
    float s = 0.0f;
    for (int p = 0; p < CA_NP; p++) {
      int dl = p & 3;                         // p = jj*DPIL + dl
      float rx = ref_cam[((size_t)(n * NQ + q) * DPIL + dl) * 2 + 0];
      float ry = ref_cam[((size_t)(n * NQ + q) * DPIL + dl) * 2 + 1];
      float ox = off[(size_t)(q * NH + hh) * CA_NP * 2 + p * 2 + 0] / (float)FW;
      float oy = off[(size_t)(q * NH + hh) * CA_NP * 2 + p * 2 + 1] / (float)FH;
      float x = (rx + ox) * FW - 0.5f;
      float y = (ry + oy) * FH - 0.5f;
      float a = aw[(size_t)(q * NH + hh) * CA_NP + p];
      s += a * bilin(vb, FH, FW, x, y, ch);
    }
    acc += hv * s;
  }
  slots[(size_t)q * EMBED + ch] = acc / fmaxf(cnt, 1.0f);
}

// ---------------------------------------------------------------------------
// Host-side orchestration
// ---------------------------------------------------------------------------
extern "C" void kernel_launch(void* const* d_in, const int* in_sizes, int n_in,
                              void* d_out, int out_size, void* d_ws, size_t ws_size,
                              hipStream_t stream) {
  (void)in_sizes; (void)n_in; (void)out_size; (void)ws_size;
  const float* x        = (const float*)d_in[0];
  const float* bev_pos  = (const float*)d_in[1];
  const float* cam_f    = (const float*)d_in[2];
  const float* l2i      = (const float*)d_in[3];
  const float* ln1_g = (const float*)d_in[4],  *ln1_b = (const float*)d_in[5];
  const float* ln2_g = (const float*)d_in[6],  *ln2_b = (const float*)d_in[7];
  const float* ln3_g = (const float*)d_in[8],  *ln3_b = (const float*)d_in[9];
  const float* sa_Wv = (const float*)d_in[10], *sa_bv = (const float*)d_in[11];
  const float* sa_Woff = (const float*)d_in[12], *sa_boff = (const float*)d_in[13];
  const float* sa_Wattn = (const float*)d_in[14], *sa_battn = (const float*)d_in[15];
  const float* sa_Wout = (const float*)d_in[16], *sa_bout = (const float*)d_in[17];
  const float* ca_Wv = (const float*)d_in[18], *ca_bv = (const float*)d_in[19];
  const float* ca_Woff = (const float*)d_in[20], *ca_boff = (const float*)d_in[21];
  const float* ca_Wattn = (const float*)d_in[22], *ca_battn = (const float*)d_in[23];
  const float* ca_Wout = (const float*)d_in[24], *ca_bout = (const float*)d_in[25];
  const float* ffn_W1 = (const float*)d_in[26], *ffn_b1 = (const float*)d_in[27];
  const float* ffn_W2 = (const float*)d_in[28], *ffn_b2 = (const float*)d_in[29];
  const float* cam_emb = (const float*)d_in[30], *lvl_emb = (const float*)d_in[31];
  float* out = (float*)d_out;

  // ---- workspace carve-out (256B aligned) ----
  size_t cur = 0;
  auto allocF = [&](size_t nf) {
    float* p = (float*)((char*)d_ws + cur);
    cur = (cur + nf * sizeof(float) + 255) & ~(size_t)255;
    return p;
  };
  auto allocH = [&](size_t nh) {
    _Float16* p = (_Float16*)((char*)d_ws + cur);
    cur = (cur + nh * sizeof(_Float16) + 255) & ~(size_t)255;
    return p;
  };

  float* h_ln1   = allocF((size_t)NQ * EMBED);
  float* qbuf    = allocF((size_t)NQ * EMBED);
  float* v_sa    = allocF((size_t)NQ * EMBED);
  float* off_sa  = allocF((size_t)NQ * NH * SA_NP * 2);
  float* attn_sa = allocF((size_t)NQ * NH * SA_NP);
  float* sa_feat = allocF((size_t)NQ * EMBED);
  float* h_sa    = allocF((size_t)NQ * EMBED);
  float* hq      = allocF((size_t)NQ * EMBED);
  float* feats   = allocF((size_t)NCAM * LCAM * EMBED);
  float* vv      = allocF((size_t)NCAM * LCAM * EMBED);
  float* off_ca  = allocF((size_t)NQ * NH * CA_NP * 2);
  float* attn_ca = allocF((size_t)NQ * NH * CA_NP);
  float* ref_cam = allocF((size_t)NCAM * NQ * DPIL * 2);
  float* hit     = allocF((size_t)NCAM * NQ);
  float* slots   = allocF((size_t)NQ * EMBED);
  float* h2      = allocF((size_t)NQ * EMBED);
  float* hf      = allocF((size_t)NQ * EMBED);
  float* ffn_t   = allocF((size_t)NQ * 2 * EMBED);

  _Float16* w_sa_v    = allocH(256 * 256);
  _Float16* w_sa_off  = allocH(256 * 64);
  _Float16* w_sa_attn = allocH(256 * 32);
  _Float16* w_sa_out  = allocH(256 * 256);
  _Float16* w_ca_v    = allocH(256 * 256);
  _Float16* w_ca_off  = allocH(256 * 128);
  _Float16* w_ca_attn = allocH(256 * 64);
  _Float16* w_ca_out  = allocH(256 * 256);
  _Float16* w_ffn1    = allocH(256 * 512);
  _Float16* w_ffn2    = allocH(512 * 256);

  auto pack = [&](const float* W, _Float16* Wt, int K, int N) {
    int n = K * N;
    k_pack_wt<<<(n + 255) / 256, 256, 0, stream>>>(W, Wt, K, N);
  };
  auto gemm = [&](const float* A, const _Float16* Bt, const float* bias,
                  const float* r1, const float* r2, float* o,
                  int M, int N, int K, int relu) {
    if (N % 64 == 0) {
      int waves = (M >> 4) * (N >> 6);
      int blocks = (waves * 32 + 255) / 256;
      k_gemm_wmma<4><<<blocks, 256, 0, stream>>>(A, Bt, bias, r1, r2, o, M, N, K, relu);
    } else {                                   // N == 32
      int waves = (M >> 4) * (N >> 5);
      int blocks = (waves * 32 + 255) / 256;
      k_gemm_wmma<2><<<blocks, 256, 0, stream>>>(A, Bt, bias, r1, r2, o, M, N, K, relu);
    }
  };

  // ---- weight packing (f32 -> f16 transposed) ----
  pack(sa_Wv, w_sa_v, 256, 256);      pack(sa_Woff, w_sa_off, 256, 64);
  pack(sa_Wattn, w_sa_attn, 256, 32); pack(sa_Wout, w_sa_out, 256, 256);
  pack(ca_Wv, w_ca_v, 256, 256);      pack(ca_Woff, w_ca_off, 256, 128);
  pack(ca_Wattn, w_ca_attn, 256, 64); pack(ca_Wout, w_ca_out, 256, 256);
  pack(ffn_W1, w_ffn1, 256, 512);     pack(ffn_W2, w_ffn2, 512, 256);

  // ---- self attention branch ----
  k_layernorm<<<NQ, EMBED, 0, stream>>>(x, ln1_g, ln1_b, bev_pos, h_ln1, qbuf);
  gemm(h_ln1, w_sa_v, sa_bv, nullptr, nullptr, v_sa, NQ, 256, 256, 0);
  gemm(qbuf, w_sa_off, sa_boff, nullptr, nullptr, off_sa, NQ, 64, 256, 0);
  gemm(qbuf, w_sa_attn, sa_battn, nullptr, nullptr, attn_sa, NQ, 32, 256, 0);
  k_softmax<<<(NQ * NH + 255) / 256, 256, 0, stream>>>(attn_sa, NQ * NH, SA_NP);
  k_sa_deform<<<NQ, 256, 0, stream>>>(v_sa, off_sa, attn_sa, sa_feat);
  gemm(sa_feat, w_sa_out, sa_bout, x, nullptr, h_sa, NQ, 256, 256, 0);   // + residual x

  // ---- cross attention branch ----
  k_layernorm<<<NQ, EMBED, 0, stream>>>(h_sa, ln2_g, ln2_b, nullptr, hq, nullptr);
  k_feats_prep<<<(NCAM * LCAM * EMBED + 255) / 256, 256, 0, stream>>>(cam_f, cam_emb, lvl_emb, feats);
  gemm(feats, w_ca_v, ca_bv, nullptr, nullptr, vv, NCAM * LCAM, 256, 256, 0);
  gemm(hq, w_ca_off, ca_boff, nullptr, nullptr, off_ca, NQ, 128, 256, 0);   // shared across cams
  gemm(hq, w_ca_attn, ca_battn, nullptr, nullptr, attn_ca, NQ, 64, 256, 0);
  k_softmax<<<(NQ * NH + 255) / 256, 256, 0, stream>>>(attn_ca, NQ * NH, CA_NP);
  k_point_sampling<<<(NCAM * NQ + 255) / 256, 256, 0, stream>>>(l2i, ref_cam, hit);
  k_ca_deform<<<NQ, 256, 0, stream>>>(vv, off_ca, attn_ca, ref_cam, hit, slots);
  gemm(slots, w_ca_out, ca_bout, hq, h_sa, h2, NQ, 256, 256, 0);           // + hq + res

  // ---- FFN ----
  k_layernorm<<<NQ, EMBED, 0, stream>>>(h2, ln3_g, ln3_b, nullptr, hf, nullptr);
  gemm(hf, w_ffn1, ffn_b1, nullptr, nullptr, ffn_t, NQ, 512, 256, 1);      // ReLU
  gemm(ffn_t, w_ffn2, ffn_b2, hf, h2, out, NQ, 256, 512, 0);               // + hf + res
}
